// NeuronSarvamDecoderLayer_23235773071427
// MI455X (gfx1250) — compile-verified, matrix operand order checked by reading, not run on hardware
//
#include <hip/hip_runtime.h>

// ---------------------------------------------------------------------------
// Types / helpers
// ---------------------------------------------------------------------------
typedef __attribute__((ext_vector_type(16))) __bf16 v16bf;
typedef __attribute__((ext_vector_type(8)))  __bf16 v8bf;
typedef __attribute__((ext_vector_type(8)))  float  v8f;

__device__ __forceinline__ __bf16 f2bf(float f) {
    unsigned u = __builtin_bit_cast(unsigned, f);
    u += 0x7fffu + ((u >> 16) & 1u);                // round-to-nearest-even
    unsigned short s = (unsigned short)(u >> 16);
    return __builtin_bit_cast(__bf16, s);
}
__device__ __forceinline__ float bf2f(__bf16 b) {
    unsigned short s = __builtin_bit_cast(unsigned short, b);
    unsigned u = ((unsigned)s) << 16;
    return __builtin_bit_cast(float, u);
}
__device__ __forceinline__ v8f wmma_bf16(v16bf a, v16bf b, v8f c) {
    return __builtin_amdgcn_wmma_f32_16x16x32_bf16(false, a, false, b, (short)0, c, false, false);
}
__device__ __forceinline__ v16bf cat16(v8bf lo, v8bf hi) {
    return __builtin_shufflevector(lo, hi, 0,1,2,3,4,5,6,7,8,9,10,11,12,13,14,15);
}

#define T_  1024
#define H_  2048
#define NH_ 32
#define NKV_ 4
#define HD_ 64
#define E_  16
#define I_  1024
#define SI_ 1024
#define EPS_ 1e-6f

#define TM 128
#define TN 64
#define TK 32

// ---------------------------------------------------------------------------
// Weight pack: fp32 [K,N] row-major -> bf16 WMMA-B-fragment layout:
//   flat[((kc*(N/16)+nc)*32 + lane)*16 + e], e-ordering per ISA 16-bit B layout
// ---------------------------------------------------------------------------
__global__ __launch_bounds__(256) void pack_b_kernel(
    const float* __restrict__ B, __bf16* __restrict__ out,
    int K, int N, long long strideIn, long long strideOut)
{
    int eg = blockIdx.z;
    const float* Bp = B + (long long)eg * strideIn;
    __bf16* op = out + (long long)eg * strideOut;
    int idx = blockIdx.x * 256 + threadIdx.x;
    int nc16 = N >> 4;
    int total = (K >> 5) * nc16 * 32;
    if (idx >= total) return;
    int lane = idx & 31;
    int rest = idx >> 5;
    int nc = rest % nc16;
    int kc = rest / nc16;
    int n = nc * 16 + (lane & 15);
    int half8 = (lane >> 4) * 8;
    v16bf t;
#pragma unroll
    for (int e2 = 0; e2 < 16; ++e2) {
        int k = kc * 32 + ((e2 < 8) ? e2 : e2 + 8) + half8;
        t[e2] = f2bf(Bp[(size_t)k * N + n]);
    }
    *(v16bf*)(op + (size_t)idx * 16) = t;
}

// ---------------------------------------------------------------------------
// GEMM (optionally grouped/gathered): C_f32 = rowScale * (A_bf16 @ Bpacked) + res
// A: bf16 row-major [M,K]; each lane streams 2 rows (2 fragments per wave).
// B: packed bf16; fragment = 32B contiguous per lane, 4 subtiles at +s*512.
// Tile 128x64, 4 waves, 8 WMMA per K-chunk, branch-free hot loop.
// ---------------------------------------------------------------------------
__global__ __launch_bounds__(128) void gemm_wmma_kernel(
    const __bf16* __restrict__ A, int lda,
    const __bf16* __restrict__ Bp, long long strideB,
    float* __restrict__ C, int ldc,
    int M, int N, int K,
    const float* __restrict__ residual,
    const float* __restrict__ rowScale,
    const int* __restrict__ grpCnt, const int* __restrict__ grpOfs,
    const int* __restrict__ tokList, const int* __restrict__ outList,
    int aOffset)
{
    const int eg = blockIdx.z;
    int m = M, base = 0;
    if (grpCnt) { m = grpCnt[eg]; base = grpOfs[eg]; }
    const __bf16* Bq = Bp + (long long)eg * strideB;
    const __bf16* Ap = A;
    if (aOffset) Ap += (size_t)base * (size_t)lda;

    const int tileM = blockIdx.y * TM;
    const int tileN = blockIdx.x * TN;
    if (tileM >= m) return;

    const int tid   = threadIdx.x;
    const int lane  = tid & 31;
    const int wv    = tid >> 5;
    const int mr    = lane & 15;
    const int half8 = (lane >> 4) * 8;
    const int nc16  = N >> 4;
    const int nBase = tileN >> 4;

    // two A row streams per wave; invalid rows clamped to m-1 (stores guarded)
    const __bf16* Arow[2];
#pragma unroll
    for (int f = 0; f < 2; ++f) {
        int arl = tileM + wv * 32 + f * 16 + mr;
        if (arl >= m) arl = m - 1;
        int ar = tokList ? tokList[base + arl] : arl;
        Arow[f] = Ap + (size_t)ar * lda + half8;
    }

    const v8f vz = {0.f,0.f,0.f,0.f,0.f,0.f,0.f,0.f};
    v8f acc[2][4];
#pragma unroll
    for (int f = 0; f < 2; ++f)
#pragma unroll
        for (int s = 0; s < 4; ++s) acc[f][s] = vz;

    const __bf16* bptr = Bq + ((size_t)nBase * 32 + lane) * 16;   // kc = 0
    const size_t bStep = (size_t)nc16 * 512;                       // elems per kc

    if (nBase + 4 <= nc16) {
        // full-tile path: branch-free inner loop
        for (int kk = 0; kk < K; kk += TK) {
            v16bf a0 = cat16(*(const v8bf*)(Arow[0] + kk),
                             *(const v8bf*)(Arow[0] + kk + 16));
            v16bf a1 = cat16(*(const v8bf*)(Arow[1] + kk),
                             *(const v8bf*)(Arow[1] + kk + 16));
            __builtin_prefetch(Arow[0] + kk + TK, 0, 1);
            __builtin_prefetch(Arow[1] + kk + TK, 0, 1);
#pragma unroll
            for (int s = 0; s < 4; ++s) {
                v16bf b = *(const v16bf*)(bptr + s * 512);
                acc[0][s] = wmma_bf16(a0, b, acc[0][s]);
                acc[1][s] = wmma_bf16(a1, b, acc[1][s]);
            }
            __builtin_prefetch(bptr + bStep, 0, 1);
            bptr += bStep;
        }
    } else {
        // edge path (N not covering 4 subtiles; only the tiny router GEMM)
        const int sMax = nc16 - nBase;  // >= 1
        for (int kk = 0; kk < K; kk += TK) {
            v16bf a0 = cat16(*(const v8bf*)(Arow[0] + kk),
                             *(const v8bf*)(Arow[0] + kk + 16));
            v16bf a1 = cat16(*(const v8bf*)(Arow[1] + kk),
                             *(const v8bf*)(Arow[1] + kk + 16));
            for (int s = 0; s < sMax; ++s) {
                v16bf b = *(const v16bf*)(bptr + s * 512);
                acc[0][s] = wmma_bf16(a0, b, acc[0][s]);
                acc[1][s] = wmma_bf16(a1, b, acc[1][s]);
            }
            bptr += bStep;
        }
    }

    // store; C/D layout: VGPR r -> M=r (lanes 0-15) / M=r+8 (lanes 16-31)
    const int sMaxS = (nc16 - nBase) < 4 ? (nc16 - nBase) : 4;
#pragma unroll
    for (int f = 0; f < 2; ++f) {
#pragma unroll
        for (int r = 0; r < 8; ++r) {
            int lrow = tileM + wv * 32 + f * 16 + r + half8;
            if (lrow < m) {
                int orow = outList ? outList[base + lrow]
                                   : (grpCnt ? base + lrow : lrow);
                float scl = rowScale ? rowScale[orow] : 1.f;
                float* crow = C + (size_t)orow * ldc;
                const float* rrow = residual ? residual + (size_t)orow * ldc : nullptr;
                for (int s = 0; s < sMaxS; ++s) {
                    int col = tileN + s * 16 + mr;
                    float vv = acc[f][s][r] * scl;
                    if (rrow) vv += rrow[col];
                    crow[col] = vv;
                }
            }
        }
    }
}

// ---------------------------------------------------------------------------
// RMSNorm: one block per row, bf16 output
// ---------------------------------------------------------------------------
__global__ __launch_bounds__(256) void rmsnorm_kernel(
    const float* __restrict__ x, const float* __restrict__ w,
    __bf16* __restrict__ out, int Hdim)
{
    __shared__ float red[256];
    int row = blockIdx.x;
    const float* xr = x + (size_t)row * Hdim;
    float s = 0.f;
    for (int j = threadIdx.x; j < Hdim; j += 256) { float v = xr[j]; s += v * v; }
    red[threadIdx.x] = s;
    __syncthreads();
    for (int o = 128; o > 0; o >>= 1) {
        if (threadIdx.x < o) red[threadIdx.x] += red[threadIdx.x + o];
        __syncthreads();
    }
    float rs = rsqrtf(red[0] / (float)Hdim + EPS_);
    for (int j = threadIdx.x; j < Hdim; j += 256)
        out[(size_t)row * Hdim + j] = f2bf(w[j] * xr[j] * rs);
}

// ---------------------------------------------------------------------------
// q/k per-head RMSNorm + RoPE; fp32 in, bf16 out. 1 wave per (pos, head).
// ---------------------------------------------------------------------------
__global__ __launch_bounds__(32) void qk_rope_kernel(
    const float* __restrict__ qf, const float* __restrict__ kf,
    __bf16* __restrict__ qo, __bf16* __restrict__ ko,
    const float* __restrict__ qw, const float* __restrict__ kw)
{
    int pos = blockIdx.x;
    int hy  = blockIdx.y;
    int d   = threadIdx.x; // 0..31
    const float* src; __bf16* dst; const float* w;
    if (hy < NH_) {
        src = qf + ((size_t)pos * NH_ + hy) * HD_;
        dst = qo + ((size_t)pos * NH_ + hy) * HD_;
        w = qw;
    } else {
        src = kf + ((size_t)pos * NKV_ + (hy - NH_)) * HD_;
        dst = ko + ((size_t)pos * NKV_ + (hy - NH_)) * HD_;
        w = kw;
    }
    float x1 = src[d], x2 = src[d + 32];
    float ss = x1 * x1 + x2 * x2;
#pragma unroll
    for (int o = 16; o > 0; o >>= 1) ss += __shfl_xor(ss, o, 32);
    float rs = rsqrtf(ss / (float)HD_ + EPS_);
    float n1 = x1 * rs * w[d], n2 = x2 * rs * w[d + 32];
    float inv = powf(8000000.0f, -(float)d / 32.0f);
    float ang = (float)pos * inv;
    float c = cosf(ang), sn = sinf(ang);
    dst[d]      = f2bf(n1 * c - n2 * sn);
    dst[d + 32] = f2bf(n2 * c + n1 * sn);
}

// v fp32 [T][NKV][HD] -> bf16 transposed [NKV][HD][T]
__global__ void vpack_kernel(const float* __restrict__ v, __bf16* __restrict__ vt)
{
    int i = blockIdx.x * 256 + threadIdx.x;
    if (i >= T_ * NKV_ * HD_) return;
    int d = i & 63;
    int r = i >> 6;
    int kv = r & (NKV_ - 1);
    int t = r >> 2;
    vt[((size_t)kv * HD_ + d) * T_ + t] = f2bf(v[i]);
}

// ---------------------------------------------------------------------------
// Causal GQA attention, WMMA. One wave per (16-query tile, head).
// q/k bf16 row-major, v bf16 transposed; all fragments are wide b128 loads.
// ---------------------------------------------------------------------------
__global__ __launch_bounds__(32) void attn_wmma_kernel(
    const __bf16* __restrict__ q, const __bf16* __restrict__ k,
    const __bf16* __restrict__ vt, __bf16* __restrict__ ctx)
{
    __shared__ __bf16 probs[16 * 1024];
    const int h    = blockIdx.y;
    const int qt   = blockIdx.x;
    const int lane = threadIdx.x;
    const int mr    = lane & 15;
    const int half8 = (lane >> 4) * 8;
    const int kvh   = h >> 3;            // NH/NKV = 8
    const int nkt   = qt + 1;            // causal
    const float scale = 0.125f;          // 1/sqrt(64)
    const __bf16 z0 = f2bf(0.f);
    const v8f vz = {0.f,0.f,0.f,0.f,0.f,0.f,0.f,0.f};

    // q fragments for the two 32-wide d chunks
    v16bf aq[2];
#pragma unroll
    for (int c = 0; c < 2; ++c) {
        const __bf16* p0 = q + ((size_t)(qt * 16 + mr) * NH_ + h) * HD_ + half8 + 32 * c;
        aq[c] = cat16(*(const v8bf*)p0, *(const v8bf*)(p0 + 16));
    }

    // pass 1: score tiles -> LDS (bf16)
    for (int kt = 0; kt < nkt; ++kt) {
        v8f sc = vz;
#pragma unroll
        for (int c = 0; c < 2; ++c) {
            const __bf16* p0 = k + ((size_t)(kt * 16 + mr) * NKV_ + kvh) * HD_ + half8 + 32 * c;
            v16bf b = cat16(*(const v8bf*)p0, *(const v8bf*)(p0 + 16));
            sc = wmma_bf16(aq[c], b, sc);
        }
#pragma unroll
        for (int r = 0; r < 8; ++r)
            probs[(r + half8) * 1024 + kt * 16 + mr] = f2bf(sc[r] * scale);
    }
    __syncthreads();

    // pass 2: per-row softmax with causal bound; zero-pad to 32-key chunks
    int pad = ((nkt * 16 + 31) / 32) * 32;
    if (lane < 16) {
        int row  = lane;
        int kend = qt * 16 + row + 1;
        float mx = -1e30f;
        for (int j = 0; j < kend; ++j) mx = fmaxf(mx, bf2f(probs[row * 1024 + j]));
        float sum = 0.f;
        for (int j = 0; j < kend; ++j) sum += __expf(bf2f(probs[row * 1024 + j]) - mx);
        float inv = 1.0f / sum;
        for (int j = 0; j < kend; ++j)
            probs[row * 1024 + j] = f2bf(__expf(bf2f(probs[row * 1024 + j]) - mx) * inv);
        for (int j = kend; j < pad; ++j) probs[row * 1024 + j] = z0;
    }
    __syncthreads();

    // pass 3: ctx = P @ V
    int nck = pad / 32;
    v8f acc[4];
#pragma unroll
    for (int s = 0; s < 4; ++s) acc[s] = vz;
    for (int ck = 0; ck < nck; ++ck) {
        const __bf16* pp = &probs[mr * 1024 + half8 + 32 * ck];
        v16bf a = cat16(*(const v8bf*)pp, *(const v8bf*)(pp + 16));
#pragma unroll
        for (int s = 0; s < 4; ++s) {
            const __bf16* pv = vt + ((size_t)kvh * HD_ + s * 16 + mr) * T_ + half8 + 32 * ck;
            v16bf b = cat16(*(const v8bf*)pv, *(const v8bf*)(pv + 16));
            acc[s] = wmma_bf16(a, b, acc[s]);
        }
    }
#pragma unroll
    for (int s = 0; s < 4; ++s)
#pragma unroll
        for (int r = 0; r < 8; ++r) {
            int qg = qt * 16 + r + half8;
            ctx[(size_t)qg * (NH_ * HD_) + h * HD_ + s * 16 + mr] = f2bf(acc[s][r]);
        }
}

// ---------------------------------------------------------------------------
// Router: sigmoid, +bias, top-4, normalized affinities * RSF
// ---------------------------------------------------------------------------
__global__ void router_topk_kernel(const float* __restrict__ logits,
                                   const float* __restrict__ bias,
                                   int* __restrict__ tkidx,
                                   float* __restrict__ affs, int T)
{
    int t = blockIdx.x * blockDim.x + threadIdx.x;
    if (t >= T) return;
    float sr[E_], sel[E_];
    bool used[E_];
    for (int e = 0; e < E_; ++e) {
        sr[e]  = 1.f / (1.f + __expf(-logits[t * E_ + e]));
        sel[e] = sr[e] + bias[e];
        used[e] = false;
    }
    int ids[4]; float av[4]; float sum = 0.f;
    for (int kq = 0; kq < 4; ++kq) {
        int bi = 0; float bv = -1e30f;
        for (int e = 0; e < E_; ++e)
            if (!used[e] && sel[e] > bv) { bv = sel[e]; bi = e; }
        used[bi] = true; ids[kq] = bi; av[kq] = sr[bi]; sum += av[kq];
    }
    float inv = 2.5f / sum;   // routed_scaling_factor folded in
    for (int kq = 0; kq < 4; ++kq) {
        tkidx[t * 4 + kq] = ids[kq];
        affs[t * 4 + kq]  = av[kq] * inv;
    }
}

// Deterministic per-expert gather lists (serial, single thread)
__global__ void build_lists_kernel(const int* __restrict__ tkidx,
                                   int* __restrict__ cnt, int* __restrict__ ofs,
                                   int* __restrict__ tokList, int* __restrict__ outList,
                                   int T)
{
    if (threadIdx.x != 0 || blockIdx.x != 0) return;
    int c[E_];
    for (int e = 0; e < E_; ++e) c[e] = 0;
    for (int i = 0; i < T * 4; ++i) c[tkidx[i]]++;
    int o = 0;
    for (int e = 0; e < E_; ++e) { ofs[e] = o; cnt[e] = c[e]; o += c[e]; c[e] = 0; }
    ofs[E_] = o;
    for (int t = 0; t < T; ++t)
        for (int kq = 0; kq < 4; ++kq) {
            int e = tkidx[t * 4 + kq];
            int p = ofs[e] + c[e]++;
            tokList[p] = t;
            outList[p] = t * 4 + kq;
        }
}

// silu(g)*u -> bf16
__global__ void silu_mul_kernel(const float* __restrict__ g, const float* __restrict__ u,
                                __bf16* __restrict__ o, long long n)
{
    long long i = blockIdx.x * (long long)blockDim.x + threadIdx.x;
    if (i >= n) return;
    float gv = g[i];
    o[i] = f2bf((gv / (1.f + __expf(-gv))) * u[i]);
}

__global__ void combine_kernel(const float* __restrict__ xattn,
                               const float* __restrict__ shrd,
                               const float* __restrict__ ybuf,
                               float* __restrict__ out, long long n)
{
    long long i = blockIdx.x * (long long)blockDim.x + threadIdx.x;
    if (i >= n) return;
    long long t = i / H_;
    int col = (int)(i % H_);
    float acc = xattn[i] + shrd[i];
#pragma unroll
    for (int kq = 0; kq < 4; ++kq)
        acc += ybuf[(size_t)(t * 4 + kq) * H_ + col];
    out[i] = acc;
}

// ---------------------------------------------------------------------------
// Host launcher
// ---------------------------------------------------------------------------
static void launch_gemm(hipStream_t st,
                        const __bf16* A, int lda, const __bf16* Bp, long long sB,
                        float* C, int ldc, int M, int N, int K,
                        const float* res, const float* rs,
                        const int* cnt, const int* ofs,
                        const int* tl, const int* ol, int aOff, int groups)
{
    dim3 g((N + TN - 1) / TN, (M + TM - 1) / TM, groups);
    gemm_wmma_kernel<<<g, 128, 0, st>>>(A, lda, Bp, sB, C, ldc, M, N, K,
                                        res, rs, cnt, ofs, tl, ol, aOff);
}

extern "C" void kernel_launch(void* const* d_in, const int* in_sizes, int n_in,
                              void* d_out, int out_size, void* d_ws, size_t ws_size,
                              hipStream_t stream)
{
    const float* x      = (const float*)d_in[0];
    const float* ln1    = (const float*)d_in[1];
    const float* ln2    = (const float*)d_in[2];
    const float* qln    = (const float*)d_in[3];
    const float* kln    = (const float*)d_in[4];
    const float* wq     = (const float*)d_in[5];
    const float* wk     = (const float*)d_in[6];
    const float* wv     = (const float*)d_in[7];
    const float* wo     = (const float*)d_in[8];
    const float* rtw    = (const float*)d_in[9];
    const float* ebias  = (const float*)d_in[10];
    const float* wg     = (const float*)d_in[11];
    const float* wu     = (const float*)d_in[12];
    const float* wd     = (const float*)d_in[13];
    const float* swg    = (const float*)d_in[14];
    const float* swu    = (const float*)d_in[15];
    const float* swd    = (const float*)d_in[16];

    char* wsp = (char*)d_ws;
    size_t off = 0;
    auto alloc = [&](size_t bytes) -> void* {
        void* p = wsp + off;
        off = (off + bytes + 255) & ~(size_t)255;
        return p;
    };

    // activations
    __bf16* h1b   = (__bf16*)alloc((size_t)T_ * H_ * 2);
    float*  qb    = (float*)alloc((size_t)T_ * NH_ * HD_ * 4);
    float*  kb    = (float*)alloc((size_t)T_ * NKV_ * HD_ * 4);
    float*  vb    = (float*)alloc((size_t)T_ * NKV_ * HD_ * 4);
    __bf16* qbb   = (__bf16*)alloc((size_t)T_ * NH_ * HD_ * 2);
    __bf16* kbb   = (__bf16*)alloc((size_t)T_ * NKV_ * HD_ * 2);
    __bf16* vt    = (__bf16*)alloc((size_t)NKV_ * HD_ * T_ * 2);
    __bf16* ctxb  = (__bf16*)alloc((size_t)T_ * H_ * 2);
    float*  xattn = (float*)alloc((size_t)T_ * H_ * 4);
    __bf16* h2b   = (__bf16*)alloc((size_t)T_ * H_ * 2);
    float*  logits= (float*)alloc((size_t)T_ * E_ * 4);
    float*  affs  = (float*)alloc((size_t)T_ * 4 * 4);
    int*    tkidx = (int*)alloc((size_t)T_ * 4 * 4);
    int*    tokL  = (int*)alloc((size_t)T_ * 4 * 4);
    int*    outL  = (int*)alloc((size_t)T_ * 4 * 4);
    int*    cnt   = (int*)alloc(E_ * 4);
    int*    ofs   = (int*)alloc((E_ + 1) * 4);
    float*  gateb = (float*)alloc((size_t)4 * T_ * I_ * 4);
    float*  upb   = (float*)alloc((size_t)4 * T_ * I_ * 4);
    __bf16* actb  = (__bf16*)alloc((size_t)4 * T_ * I_ * 2);
    float*  ybuf  = (float*)alloc((size_t)4 * T_ * H_ * 4);
    float*  sg    = (float*)alloc((size_t)T_ * SI_ * 4);
    float*  su    = (float*)alloc((size_t)T_ * SI_ * 4);
    __bf16* sab   = (__bf16*)alloc((size_t)T_ * SI_ * 2);
    float*  shrd  = (float*)alloc((size_t)T_ * H_ * 4);

    // packed bf16 weights
    __bf16* pwq  = (__bf16*)alloc((size_t)H_ * H_ * 2);
    __bf16* pwk  = (__bf16*)alloc((size_t)H_ * (NKV_ * HD_) * 2);
    __bf16* pwv  = (__bf16*)alloc((size_t)H_ * (NKV_ * HD_) * 2);
    __bf16* pwo  = (__bf16*)alloc((size_t)H_ * H_ * 2);
    __bf16* prt  = (__bf16*)alloc((size_t)H_ * E_ * 2);
    __bf16* pwg  = (__bf16*)alloc((size_t)E_ * H_ * I_ * 2);
    __bf16* pwu  = (__bf16*)alloc((size_t)E_ * H_ * I_ * 2);
    __bf16* pwd  = (__bf16*)alloc((size_t)E_ * I_ * H_ * 2);
    __bf16* pswg = (__bf16*)alloc((size_t)H_ * SI_ * 2);
    __bf16* pswu = (__bf16*)alloc((size_t)H_ * SI_ * 2);
    __bf16* pswd = (__bf16*)alloc((size_t)SI_ * H_ * 2);
    (void)ws_size; (void)n_in; (void)in_sizes; (void)out_size;

    auto pack = [&](const float* src, __bf16* dst, int K, int N,
                    long long sIn, long long sOut, int groups) {
        int total = (K >> 5) * (N >> 4) * 32;
        dim3 g((total + 255) / 256, 1, groups);
        pack_b_kernel<<<g, 256, 0, stream>>>(src, dst, K, N, sIn, sOut);
    };

    // pack all weights (amortized; each is re-read many times by the GEMMs)
    pack(wq,  pwq,  H_, H_,          0, 0, 1);
    pack(wk,  pwk,  H_, NKV_ * HD_,  0, 0, 1);
    pack(wv,  pwv,  H_, NKV_ * HD_,  0, 0, 1);
    pack(wo,  pwo,  H_, H_,          0, 0, 1);
    pack(rtw, prt,  H_, E_,          0, 0, 1);
    pack(wg,  pwg,  H_, I_, (long long)H_ * I_, (long long)H_ * I_, E_);
    pack(wu,  pwu,  H_, I_, (long long)H_ * I_, (long long)H_ * I_, E_);
    pack(wd,  pwd,  I_, H_, (long long)I_ * H_, (long long)I_ * H_, E_);
    pack(swg, pswg, H_, SI_,         0, 0, 1);
    pack(swu, pswu, H_, SI_,         0, 0, 1);
    pack(swd, pswd, SI_, H_,         0, 0, 1);

    // --- attention block ---
    rmsnorm_kernel<<<T_, 256, 0, stream>>>(x, ln1, h1b, H_);
    launch_gemm(stream, h1b, H_, pwq, 0, qb, NH_ * HD_, T_, NH_ * HD_, H_,
                nullptr, nullptr, nullptr, nullptr, nullptr, nullptr, 0, 1);
    launch_gemm(stream, h1b, H_, pwk, 0, kb, NKV_ * HD_, T_, NKV_ * HD_, H_,
                nullptr, nullptr, nullptr, nullptr, nullptr, nullptr, 0, 1);
    launch_gemm(stream, h1b, H_, pwv, 0, vb, NKV_ * HD_, T_, NKV_ * HD_, H_,
                nullptr, nullptr, nullptr, nullptr, nullptr, nullptr, 0, 1);
    qk_rope_kernel<<<dim3(T_, NH_ + NKV_), 32, 0, stream>>>(qb, kb, qbb, kbb, qln, kln);
    vpack_kernel<<<(T_ * NKV_ * HD_ + 255) / 256, 256, 0, stream>>>(vb, vt);
    attn_wmma_kernel<<<dim3(T_ / 16, NH_), 32, 0, stream>>>(qbb, kbb, vt, ctxb);
    launch_gemm(stream, ctxb, H_, pwo, 0, xattn, H_, T_, H_, H_,
                /*residual=*/x, nullptr, nullptr, nullptr, nullptr, nullptr, 0, 1);

    // --- MoE block ---
    rmsnorm_kernel<<<T_, 256, 0, stream>>>(xattn, ln2, h2b, H_);
    launch_gemm(stream, h2b, H_, prt, 0, logits, E_, T_, E_, H_,
                nullptr, nullptr, nullptr, nullptr, nullptr, nullptr, 0, 1);
    router_topk_kernel<<<(T_ + 127) / 128, 128, 0, stream>>>(logits, ebias, tkidx, affs, T_);
    build_lists_kernel<<<1, 1, 0, stream>>>(tkidx, cnt, ofs, tokL, outL, T_);

    // routed experts: gathered grouped GEMMs
    launch_gemm(stream, h2b, H_, pwg, (long long)H_ * I_, gateb, I_, T_, I_, H_,
                nullptr, nullptr, cnt, ofs, tokL, nullptr, 0, E_);
    launch_gemm(stream, h2b, H_, pwu, (long long)H_ * I_, upb, I_, T_, I_, H_,
                nullptr, nullptr, cnt, ofs, tokL, nullptr, 0, E_);
    {
        long long n = (long long)4 * T_ * I_;
        silu_mul_kernel<<<(unsigned)((n + 255) / 256), 256, 0, stream>>>(gateb, upb, actb, n);
    }
    launch_gemm(stream, actb, I_, pwd, (long long)I_ * H_, ybuf, H_, T_, H_, I_,
                nullptr, /*rowScale=*/affs, cnt, ofs, nullptr, outL, /*aOffset=*/1, E_);

    // shared expert
    launch_gemm(stream, h2b, H_, pswg, 0, sg, SI_, T_, SI_, H_,
                nullptr, nullptr, nullptr, nullptr, nullptr, nullptr, 0, 1);
    launch_gemm(stream, h2b, H_, pswu, 0, su, SI_, T_, SI_, H_,
                nullptr, nullptr, nullptr, nullptr, nullptr, nullptr, 0, 1);
    {
        long long n = (long long)T_ * SI_;
        silu_mul_kernel<<<(unsigned)((n + 255) / 256), 256, 0, stream>>>(sg, su, sab, n);
    }
    launch_gemm(stream, sab, SI_, pswd, 0, shrd, H_, T_, H_, SI_,
                nullptr, nullptr, nullptr, nullptr, nullptr, nullptr, 0, 1);

    // final combine: out = xattn + shared + sum_k ybuf[t*4+k]
    {
        long long n = (long long)T_ * H_;
        combine_kernel<<<(unsigned)((n + 255) / 256), 256, 0, stream>>>(
            xattn, shrd, ybuf, (float*)d_out, n);
    }
}
